// CSOCSSC_v48_3109556322363
// MI455X (gfx1250) — matrix operand order repaired and checked by reading, not compiled, not run on hardware
//
#include <hip/hip_runtime.h>
#include <cmath>

namespace {

constexpr int NTOK = 1024;
constexpr int CDIM = 384;
constexpr int HNUM = 12;
constexpr int CH   = 32;
constexpr int HP3  = 144;   // H*P*3
constexpr int HP3P = 160;   // padded to multiple of 32 for WMMA K loop
constexpr int KCAT = 64;    // 32 scalar dims + 12 point dims + 20 zero pad
constexpr float EPSF = 1e-5f;

typedef __attribute__((ext_vector_type(16))) __bf16 v16bf;
typedef __attribute__((ext_vector_type(8)))  float  v8f;

__device__ __forceinline__ v8f wmma_bf16(v16bf a, v16bf b, v8f c) {
  return __builtin_amdgcn_wmma_f32_16x16x32_bf16(false, a, false, b, (short)0, c,
                                                 false, false);
}

// ---- fragment loaders (wave32 WMMA, contiguous k = k0 + 16*half + s used in both A and B) ----
// A 16x32 bf16: lane<16 holds row M=lane, K in {k0..k0+15};
//               lane>=16 holds row M=lane-16, K in {k0+16..k0+31}.
__device__ __forceinline__ v16bf frag_a_bf16(const __bf16* A, int lda, int m0,
                                             int k0, int lane) {
  int half = lane >> 4, lr = lane & 15;
  const __bf16* p = A + (size_t)(m0 + lr) * lda;
  int k1 = k0 + half * 16, k2 = k0 + half * 16 + 8;
  v16bf a;
#pragma unroll
  for (int j = 0; j < 8; ++j) { a[j] = p[k1 + j]; a[8 + j] = p[k2 + j]; }
  return a;
}

__device__ __forceinline__ v16bf frag_a_f32(const float* A, int lda, int m0,
                                            int k0, int lane) {
  int half = lane >> 4, lr = lane & 15;
  const float* p = A + (size_t)(m0 + lr) * lda;
  int k1 = k0 + half * 16, k2 = k0 + half * 16 + 8;
  v16bf a;
#pragma unroll
  for (int j = 0; j < 8; ++j) {
    a[j]     = (__bf16)p[k1 + j];
    a[8 + j] = (__bf16)p[k2 + j];
  }
  return a;
}

// B 32x16: VGPR i, lane<16 -> (K=k0+i, N=n0+lane); lane>=16 -> K offset +16.
__device__ __forceinline__ v16bf frag_b_f32(const float* W, int ldw, int k0,
                                            int n0, int lane) {
  int half = lane >> 4, lr = lane & 15;
  const float* p = W + (size_t)(k0 + half * 16) * ldw + n0 + lr;
  v16bf b;
#pragma unroll
  for (int i = 0; i < 16; ++i) b[i] = (__bf16)p[(size_t)i * ldw];
  return b;
}

__device__ __forceinline__ v16bf frag_b_bf16(const __bf16* W, int ldw, int k0,
                                             int n0, int lane) {
  int half = lane >> 4, lr = lane & 15;
  const __bf16* p = W + (size_t)(k0 + half * 16) * ldw + n0 + lr;
  v16bf b;
#pragma unroll
  for (int i = 0; i < 16; ++i) b[i] = p[(size_t)i * ldw];
  return b;
}

// B = X^T where X is row-major [N x K]: b[i] = X[n0+lr][k0+half*16+i] (contiguous)
__device__ __forceinline__ v16bf frag_bT_bf16(const __bf16* X, int ldx, int k0,
                                              int n0, int lane) {
  int half = lane >> 4, lr = lane & 15;
  const __bf16* p = X + (size_t)(n0 + lr) * ldx + k0 + half * 16;
  v16bf b;
#pragma unroll
  for (int i = 0; i < 16; ++i) b[i] = p[i];
  return b;
}

// column-guarded B (for Wpb: Nout=12 < 16)
__device__ __forceinline__ v16bf frag_b_f32_gn(const float* W, int ldw, int Nact,
                                               int k0, int n0, int lane) {
  int half = lane >> 4, lr = lane & 15;
  bool ok = (n0 + lr) < Nact;
  v16bf b;
#pragma unroll
  for (int i = 0; i < 16; ++i)
    b[i] = ok ? (__bf16)W[(size_t)(k0 + half * 16 + i) * ldw + n0 + lr]
              : (__bf16)0.f;
  return b;
}

// K-guarded B (for Wpo: Kdim=144, loop padded to 160)
__device__ __forceinline__ v16bf frag_b_f32_gk(const float* W, int ldw, int Kact,
                                               int k0, int n0, int lane) {
  int half = lane >> 4, lr = lane & 15;
  v16bf b;
#pragma unroll
  for (int i = 0; i < 16; ++i) {
    int k = k0 + half * 16 + i;
    b[i] = (k < Kact) ? (__bf16)W[(size_t)k * ldw + n0 + lr] : (__bf16)0.f;
  }
  return b;
}

// ------------------------------- kernels -----------------------------------

__global__ void ipa_cvt_bf16(const float* __restrict__ x, __bf16* __restrict__ y,
                             int n) {
  int i = blockIdx.x * blockDim.x + threadIdx.x;
  if (i < n) y[i] = (__bf16)x[i];
}

__global__ void ipa_zero_f32(float* __restrict__ p, int n) {
  int i = blockIdx.x * blockDim.x + threadIdx.x;
  if (i < n) p[i] = 0.f;
}

// Out[M x Nout] = A(bf16)[M x K] @ W(f32)[K x Nout] + bias ; one wave per 16x16 tile
__global__ void ipa_gemm_bias(const __bf16* __restrict__ A,
                              const float* __restrict__ W,
                              const float* __restrict__ bias,
                              float* __restrict__ Out, int M, int Kdim, int Nout) {
  int gt = blockIdx.x * blockDim.x + threadIdx.x;
  int wave = gt >> 5, lane = gt & 31;
  int tilesN = Nout >> 4;
  int tm = wave / tilesN, tn = wave % tilesN;
  if (tm * 16 >= M) return;
  v8f acc = {};
  for (int k0 = 0; k0 < Kdim; k0 += 32) {
    v16bf a = frag_a_bf16(A, Kdim, tm * 16, k0, lane);
    v16bf b = frag_b_f32(W, Nout, k0, tn * 16, lane);
    acc = wmma_bf16(a, b, acc);
  }
  int half = lane >> 4, lr = lane & 15;
  int col = tn * 16 + lr;
  float bb = bias[col];
#pragma unroll
  for (int r = 0; r < 8; ++r)
    Out[(size_t)(tm * 16 + half * 8 + r) * Nout + col] = acc[r] + bb;
}

// Rigid transform of q/k/v points, build concatenated Q/K for fused logits GEMM,
// bf16 V and V_pts, and the -0.5*scale*(q2/k2) bias vectors. One thread per (n,h).
__global__ void ipa_points(const float* __restrict__ Q, const float* __restrict__ K,
                           const float* __restrict__ V,
                           const float* __restrict__ QPr,
                           const float* __restrict__ KPr,
                           const float* __restrict__ VPr,
                           const float* __restrict__ rot,
                           const float* __restrict__ trans,
                           __bf16* __restrict__ Qc, __bf16* __restrict__ Kc,
                           __bf16* __restrict__ Vb, __bf16* __restrict__ Vpb,
                           float* __restrict__ r2, float* __restrict__ c2,
                           float scale) {
  int idx = blockIdx.x * blockDim.x + threadIdx.x;
  if (idx >= NTOK * HNUM) return;
  int n = idx / HNUM, h = idx % HNUM;
  float R[9], T[3];
#pragma unroll
  for (int i = 0; i < 9; ++i) R[i] = rot[n * 9 + i];
#pragma unroll
  for (int e = 0; e < 3; ++e) T[e] = trans[n * 3 + e];

  const float* qp = QPr + (size_t)n * HP3 + h * 12;
  const float* kp = KPr + (size_t)n * HP3 + h * 12;
  const float* vp = VPr + (size_t)n * HP3 + h * 12;
  float qg[12], kg[12], vg[12], q2 = 0.f, k2 = 0.f;
#pragma unroll
  for (int p = 0; p < 4; ++p) {
    float q0 = qp[p * 3], q1 = qp[p * 3 + 1], q2l = qp[p * 3 + 2];
    float k0 = kp[p * 3], k1 = kp[p * 3 + 1], k2l = kp[p * 3 + 2];
    float v0 = vp[p * 3], v1 = vp[p * 3 + 1], v2l = vp[p * 3 + 2];
#pragma unroll
    for (int e = 0; e < 3; ++e) {
      float g = q0 * R[e] + q1 * R[3 + e] + q2l * R[6 + e] + T[e];
      qg[p * 3 + e] = g; q2 += g * g;
      g = k0 * R[e] + k1 * R[3 + e] + k2l * R[6 + e] + T[e];
      kg[p * 3 + e] = g; k2 += g * g;
      vg[p * 3 + e] = v0 * R[e] + v1 * R[3 + e] + v2l * R[6 + e] + T[e];
    }
  }
  __bf16* qc = Qc + (size_t)idx * KCAT;
  __bf16* kc = Kc + (size_t)idx * KCAT;
  const float* qrow = Q + (size_t)n * CDIM + h * CH;
  const float* krow = K + (size_t)n * CDIM + h * CH;
#pragma unroll
  for (int c = 0; c < CH; ++c) {
    qc[c] = (__bf16)(qrow[c] * scale);
    kc[c] = (__bf16)krow[c];
  }
#pragma unroll
  for (int j = 0; j < 12; ++j) {
    qc[CH + j] = (__bf16)(qg[j] * scale);
    kc[CH + j] = (__bf16)kg[j];
  }
#pragma unroll
  for (int j = 44; j < KCAT; ++j) { qc[j] = (__bf16)0.f; kc[j] = (__bf16)0.f; }

  __bf16* vb = Vb + (size_t)idx * CH;
  const float* vrow = V + (size_t)n * CDIM + h * CH;
#pragma unroll
  for (int c = 0; c < CH; ++c) vb[c] = (__bf16)vrow[c];
  __bf16* vpb = Vpb + (size_t)idx * 16;
#pragma unroll
  for (int j = 0; j < 12; ++j) vpb[j] = (__bf16)vg[j];
#pragma unroll
  for (int j = 12; j < 16; ++j) vpb[j] = (__bf16)0.f;

  r2[h * NTOK + n] = -0.5f * scale * q2;
  c2[h * NTOK + n] = -0.5f * scale * k2;
}

// ---------------- pair-bias kernel with async-to-LDS staging -----------------
// pb[h][m][n] = pair[m,n,:] @ Wpb[:,h] + bpb[h].
// This streams the 512 MB pair tensor (the HBM roofline term: ~22us @ 23.3TB/s),
// so it gets the CDNA5 async DMA path: each 16x128-f32 tile of pair is one
// contiguous 8KB block; GLOBAL_LOAD_ASYNC_TO_LDS_B128 moves one 512B tile-row
// per instruction (32 lanes x 16B). Double-buffered in LDS, ASYNCcnt-tracked,
// s_wait_asynccnt overlaps tile t+1 DMA with tile t WMMAs.

constexpr int PB_ROWB  = 512 + 32;        // 128 f32 + 8-dword pad (bank spread)
constexpr int PB_TILEB = 16 * PB_ROWB;    // 8704 B per tile buffer
constexpr int PB_WAVES = 2;               // waves per block (LDS budget < 64KB)

__device__ __forceinline__ void pb_issue_tile(char* ldsbuf, const float* g,
                                              int lane) {
  unsigned lbase = (unsigned)(uintptr_t)ldsbuf + lane * 16;  // LDS offset (low 32b)
  const char* gb = (const char*)g + lane * 16;
#pragma unroll
  for (int r = 0; r < 16; ++r) {
    unsigned loff = lbase + r * PB_ROWB;
    const void* ga = gb + r * 512;
    asm volatile("global_load_async_to_lds_b128 %0, %1, off"
                 :: "v"(loff), "v"(ga) : "memory");
  }
}

__device__ __forceinline__ v16bf pb_frag_a_lds(const char* tile, int k0, int lane) {
  int half = lane >> 4, lr = lane & 15;
  const float* p = (const float*)(tile + (size_t)lr * PB_ROWB);
  int k1 = k0 + half * 16, k2 = k0 + half * 16 + 8;
  v16bf a;
#pragma unroll
  for (int j = 0; j < 8; ++j) {
    a[j]     = (__bf16)p[k1 + j];
    a[8 + j] = (__bf16)p[k2 + j];
  }
  return a;
}

__global__ __launch_bounds__(PB_WAVES * 32)
void ipa_pairbias(const float* __restrict__ pair, const float* __restrict__ Wpb,
                  const float* __restrict__ bpb, float* __restrict__ pb) {
  __shared__ char lds[PB_WAVES * 2 * PB_TILEB];
  int wid  = threadIdx.x >> 5;
  int lane = threadIdx.x & 31;
  int wave = blockIdx.x * PB_WAVES + wid;
  constexpr int NQ = NTOK / 64;  // 16 n-quads (4 tiles of 16) per m
  int m  = wave / NQ;
  int nq = wave % NQ;
  if (m >= NTOK) return;
  int n0base = nq * 64;
  char* buf0 = lds + (size_t)wid * 2 * PB_TILEB;
  char* buf1 = buf0 + PB_TILEB;
  const float* gbase = pair + ((size_t)m * NTOK + n0base) * 128;

  // Wpb B-fragments (col-guarded to 12 heads), hoisted: reused by all 4 tiles
  v16bf wb0 = frag_b_f32_gn(Wpb, HNUM, HNUM, 0,  0, lane);
  v16bf wb1 = frag_b_f32_gn(Wpb, HNUM, HNUM, 32, 0, lane);
  v16bf wb2 = frag_b_f32_gn(Wpb, HNUM, HNUM, 64, 0, lane);
  v16bf wb3 = frag_b_f32_gn(Wpb, HNUM, HNUM, 96, 0, lane);

  int half = lane >> 4, lr = lane & 15;
  float bias = (lr < HNUM) ? bpb[lr] : 0.f;

  pb_issue_tile(buf0, gbase, lane);  // prime the pipeline
#pragma unroll
  for (int t = 0; t < 4; ++t) {
    if (t < 3) {
      pb_issue_tile((t & 1) ? buf0 : buf1, gbase + (size_t)(t + 1) * 16 * 128,
                    lane);
      // 32 async ops outstanding; oldest 16 (tile t) done when cnt <= 16
      asm volatile("s_wait_asynccnt 0x10" ::: "memory");
    } else {
      asm volatile("s_wait_asynccnt 0x0" ::: "memory");
    }
    const char* cur = (t & 1) ? buf1 : buf0;
    v8f acc = {};
    acc = wmma_bf16(pb_frag_a_lds(cur, 0,  lane), wb0, acc);
    acc = wmma_bf16(pb_frag_a_lds(cur, 32, lane), wb1, acc);
    acc = wmma_bf16(pb_frag_a_lds(cur, 64, lane), wb2, acc);
    acc = wmma_bf16(pb_frag_a_lds(cur, 96, lane), wb3, acc);
    if (lr < HNUM) {
      int n0 = n0base + t * 16;
      size_t base = (size_t)lr * NTOK * NTOK + (size_t)m * NTOK + n0 + half * 8;
#pragma unroll
      for (int r = 0; r < 8; ++r) pb[base + r] = acc[r] + bias;
    }
  }
}

// logits (in place over pb buffer): fused scalar-QK + point-QK (K=64 padded),
// plus q2/k2 bias vectors and mask. One wave per (h, 16 m-rows, 256 n-cols).
__global__ void ipa_logits(const __bf16* __restrict__ Qc,
                           const __bf16* __restrict__ Kc,
                           const float* __restrict__ r2,
                           const float* __restrict__ c2,
                           const unsigned char* __restrict__ mask,
                           float* __restrict__ att) {
  int gt = blockIdx.x * blockDim.x + threadIdx.x;
  int wave = gt >> 5, lane = gt & 31;
  constexpr int MT = NTOK / 16, NQ = 4;  // 64 m-tiles, 4 n-quadrants
  int h = wave / (MT * NQ);
  if (h >= HNUM) return;
  int rem = wave % (MT * NQ);
  int m0 = (rem / NQ) * 16;
  int nbase = (rem % NQ) * 256;
  int half = lane >> 4, lr = lane & 15;

  const __bf16* Qh = Qc + (size_t)h * KCAT;
  const __bf16* Kh = Kc + (size_t)h * KCAT;
  v16bf qa0 = frag_a_bf16(Qh, HNUM * KCAT, m0, 0, lane);
  v16bf qa1 = frag_a_bf16(Qh, HNUM * KCAT, m0, 32, lane);

  float r2v[8];
  bool mrow[8];
#pragma unroll
  for (int r = 0; r < 8; ++r) {
    int mi = m0 + half * 8 + r;
    r2v[r] = r2[h * NTOK + mi];
    mrow[r] = mask[mi] != 0;
  }

  for (int t = 0; t < 16; ++t) {
    int n0 = nbase + t * 16;
    v16bf b0 = frag_bT_bf16(Kh, HNUM * KCAT, 0, n0, lane);
    v16bf b1 = frag_bT_bf16(Kh, HNUM * KCAT, 32, n0, lane);
    v8f acc = {};
    acc = wmma_bf16(qa0, b0, acc);
    acc = wmma_bf16(qa1, b1, acc);
    int ncol = n0 + lr;
    float cb = c2[h * NTOK + ncol];
    bool mc = mask[ncol] != 0;
    size_t base = (size_t)h * NTOK * NTOK + (size_t)(m0 + half * 8) * NTOK + ncol;
#pragma unroll
    for (int r = 0; r < 8; ++r) {
      size_t idx = base + (size_t)r * NTOK;
      float v = att[idx] + acc[r] + r2v[r] + cb;
      att[idx] = (mrow[r] && mc) ? v : -1e9f;
    }
  }
}

// row softmax over N=1024, in place; one 256-thread block per (h,m) row.
// The [12,1024,1024] tensor fits in MI455X's 192MB L2, so this pass stays on-chip.
__global__ void ipa_softmax(float* __restrict__ att) {
  __shared__ float red[256];
  float* p = att + (size_t)blockIdx.x * NTOK;
  int t = threadIdx.x;
  float lm = -1e30f;
  for (int i = t; i < NTOK; i += 256) lm = fmaxf(lm, p[i]);
  red[t] = lm;
  __syncthreads();
  for (int s = 128; s > 0; s >>= 1) {
    if (t < s) red[t] = fmaxf(red[t], red[t + s]);
    __syncthreads();
  }
  float mx = red[0];
  __syncthreads();
  float ls = 0.f;
  for (int i = t; i < NTOK; i += 256) {
    float e = __expf(p[i] - mx);
    p[i] = e;
    ls += e;
  }
  red[t] = ls;
  __syncthreads();
  for (int s = 128; s > 0; s >>= 1) {
    if (t < s) red[t] += red[t + s];
    __syncthreads();
  }
  float inv = 1.f / red[0];
  for (int i = t; i < NTOK; i += 256) p[i] *= inv;
}

// AV: one wave per (h, 16 m-rows); accumulates attn@V (16x32) and attn@Vpts (16x16)
// over K=1024 in 32 WMMA steps x3.
__global__ void ipa_av(const float* __restrict__ att, const __bf16* __restrict__ Vb,
                       const __bf16* __restrict__ Vpb, float* __restrict__ OS,
                       float* __restrict__ OP) {
  int gt = blockIdx.x * blockDim.x + threadIdx.x;
  int wave = gt >> 5, lane = gt & 31;
  int h = wave / (NTOK / 16);
  if (h >= HNUM) return;
  int m0 = (wave % (NTOK / 16)) * 16;
  const float* A = att + (size_t)h * NTOK * NTOK;
  const __bf16* Vh = Vb + (size_t)h * CH;
  const __bf16* Vph = Vpb + (size_t)h * 16;
  v8f a0 = {}, a1 = {}, ap = {};
  for (int k0 = 0; k0 < NTOK; k0 += 32) {
    v16bf a = frag_a_f32(A, NTOK, m0, k0, lane);
    v16bf b0 = frag_b_bf16(Vh, HNUM * CH, k0, 0, lane);
    v16bf b1 = frag_b_bf16(Vh, HNUM * CH, k0, 16, lane);
    v16bf bp = frag_b_bf16(Vph, HNUM * 16, k0, 0, lane);
    a0 = wmma_bf16(a, b0, a0);
    a1 = wmma_bf16(a, b1, a1);
    ap = wmma_bf16(a, bp, ap);
  }
  int half = lane >> 4, lr = lane & 15;
#pragma unroll
  for (int r = 0; r < 8; ++r) {
    int m = m0 + half * 8 + r;
    OS[(size_t)m * CDIM + h * CH + lr] = a0[r];
    OS[(size_t)m * CDIM + h * CH + 16 + lr] = a1[r];
    if (lr < 12) OP[(size_t)m * HP3P + h * 12 + lr] = ap[r];
  }
}

// x = single + OS@Wo + OP@Wpo + bo + bpo ; one wave per 16x16 tile of [1024x384]
__global__ void ipa_outproj(const float* __restrict__ OS, const float* __restrict__ Wo,
                            const float* __restrict__ bo, const float* __restrict__ OP,
                            const float* __restrict__ Wpo, const float* __restrict__ bpo,
                            const float* __restrict__ single, float* __restrict__ xb) {
  int gt = blockIdx.x * blockDim.x + threadIdx.x;
  int wave = gt >> 5, lane = gt & 31;
  constexpr int TN = CDIM / 16;  // 24
  int tm = wave / TN, tn = wave % TN;
  if (tm * 16 >= NTOK) return;
  v8f acc = {};
  for (int k0 = 0; k0 < CDIM; k0 += 32) {
    v16bf a = frag_a_f32(OS, CDIM, tm * 16, k0, lane);
    v16bf b = frag_b_f32(Wo, CDIM, k0, tn * 16, lane);
    acc = wmma_bf16(a, b, acc);
  }
  for (int k0 = 0; k0 < HP3P; k0 += 32) {  // OP zero-padded beyond 144
    v16bf a = frag_a_f32(OP, HP3P, tm * 16, k0, lane);
    v16bf b = frag_b_f32_gk(Wpo, CDIM, HP3, k0, tn * 16, lane);
    acc = wmma_bf16(a, b, acc);
  }
  int half = lane >> 4, lr = lane & 15;
  int col = tn * 16 + lr;
  float bb = bo[col] + bpo[col];
#pragma unroll
  for (int r = 0; r < 8; ++r) {
    size_t i = (size_t)(tm * 16 + half * 8 + r) * CDIM + col;
    xb[i] = acc[r] + bb + single[i];
  }
}

// LayerNorm per row; one 128-thread block per token.
__global__ void ipa_layernorm(const float* __restrict__ x,
                              const float* __restrict__ gamma,
                              const float* __restrict__ beta,
                              float* __restrict__ out) {
  __shared__ float red[128];
  const float* p = x + (size_t)blockIdx.x * CDIM;
  float* o = out + (size_t)blockIdx.x * CDIM;
  int t = threadIdx.x;
  float s = 0.f;
  for (int i = t; i < CDIM; i += 128) s += p[i];
  red[t] = s;
  __syncthreads();
  for (int w = 64; w > 0; w >>= 1) {
    if (t < w) red[t] += red[t + w];
    __syncthreads();
  }
  float mu = red[0] / CDIM;
  __syncthreads();
  float v = 0.f;
  for (int i = t; i < CDIM; i += 128) {
    float d = p[i] - mu;
    v += d * d;
  }
  red[t] = v;
  __syncthreads();
  for (int w = 64; w > 0; w >>= 1) {
    if (t < w) red[t] += red[t + w];
    __syncthreads();
  }
  float rstd = rsqrtf(red[0] / CDIM + EPSF);
  for (int i = t; i < CDIM; i += 128) o[i] = (p[i] - mu) * rstd * gamma[i] + beta[i];
}

}  // namespace

extern "C" void kernel_launch(void* const* d_in, const int* in_sizes, int n_in,
                              void* d_out, int out_size, void* d_ws, size_t ws_size,
                              hipStream_t stream) {
  const float* single = (const float*)d_in[0];
  const float* pair   = (const float*)d_in[1];
  const float* rot    = (const float*)d_in[2];
  const float* trans  = (const float*)d_in[3];
  const unsigned char* mask = (const unsigned char*)d_in[4];
  const float* Wq = (const float*)d_in[5],  *bq  = (const float*)d_in[6];
  const float* Wk = (const float*)d_in[7],  *bk  = (const float*)d_in[8];
  const float* Wv = (const float*)d_in[9],  *bv  = (const float*)d_in[10];
  const float* Wpb = (const float*)d_in[11], *bpb = (const float*)d_in[12];
  const float* Wqp = (const float*)d_in[13], *bqp = (const float*)d_in[14];
  const float* Wkp = (const float*)d_in[15], *bkp = (const float*)d_in[16];
  const float* Wvp = (const float*)d_in[17], *bvp = (const float*)d_in[18];
  const float* Wo  = (const float*)d_in[19], *bo  = (const float*)d_in[20];
  const float* Wpo = (const float*)d_in[21], *bpo = (const float*)d_in[22];
  const float* gamma = (const float*)d_in[23], *beta = (const float*)d_in[24];
  float* out = (float*)d_out;

  char* w = (char*)d_ws;
  auto alloc = [&](size_t bytes) -> void* {
    void* p = (void*)w;
    w += (bytes + 255) & ~(size_t)255;
    return p;
  };
  __bf16* sbf  = (__bf16*)alloc((size_t)NTOK * CDIM * 2);
  float* Q     = (float*)alloc((size_t)NTOK * CDIM * 4);
  float* K     = (float*)alloc((size_t)NTOK * CDIM * 4);
  float* V     = (float*)alloc((size_t)NTOK * CDIM * 4);
  float* QPr   = (float*)alloc((size_t)NTOK * HP3 * 4);
  float* KPr   = (float*)alloc((size_t)NTOK * HP3 * 4);
  float* VPr   = (float*)alloc((size_t)NTOK * HP3 * 4);
  __bf16* Qc   = (__bf16*)alloc((size_t)NTOK * HNUM * KCAT * 2);
  __bf16* Kc   = (__bf16*)alloc((size_t)NTOK * HNUM * KCAT * 2);
  __bf16* Vb   = (__bf16*)alloc((size_t)NTOK * HNUM * CH * 2);
  __bf16* Vpb  = (__bf16*)alloc((size_t)NTOK * HNUM * 16 * 2);
  float* r2    = (float*)alloc((size_t)HNUM * NTOK * 4);
  float* c2    = (float*)alloc((size_t)HNUM * NTOK * 4);
  float* att   = (float*)alloc((size_t)HNUM * NTOK * NTOK * 4);  // pb -> logits -> attn
  float* OS    = (float*)alloc((size_t)NTOK * CDIM * 4);
  float* OP    = (float*)alloc((size_t)NTOK * HP3P * 4);
  float* xb    = (float*)alloc((size_t)NTOK * CDIM * 4);
  (void)ws_size; (void)n_in; (void)in_sizes; (void)out_size;

  const float scale = 1.0f / sqrtf((float)CH);
  auto nblk = [](long threads, int bs) { return (int)((threads + bs - 1) / bs); };

  // 1. single -> bf16
  ipa_cvt_bf16<<<nblk((long)NTOK * CDIM, 256), 256, 0, stream>>>(single, sbf,
                                                                 NTOK * CDIM);
  // 2. six projections (WMMA GEMMs)
  {
    int wv = (NTOK / 16) * (CDIM / 16);  // 1536 waves
    ipa_gemm_bias<<<nblk((long)wv * 32, 256), 256, 0, stream>>>(sbf, Wq, bq, Q, NTOK, CDIM, CDIM);
    ipa_gemm_bias<<<nblk((long)wv * 32, 256), 256, 0, stream>>>(sbf, Wk, bk, K, NTOK, CDIM, CDIM);
    ipa_gemm_bias<<<nblk((long)wv * 32, 256), 256, 0, stream>>>(sbf, Wv, bv, V, NTOK, CDIM, CDIM);
    int wvp = (NTOK / 16) * (HP3 / 16);  // 576 waves
    ipa_gemm_bias<<<nblk((long)wvp * 32, 256), 256, 0, stream>>>(sbf, Wqp, bqp, QPr, NTOK, CDIM, HP3);
    ipa_gemm_bias<<<nblk((long)wvp * 32, 256), 256, 0, stream>>>(sbf, Wkp, bkp, KPr, NTOK, CDIM, HP3);
    ipa_gemm_bias<<<nblk((long)wvp * 32, 256), 256, 0, stream>>>(sbf, Wvp, bvp, VPr, NTOK, CDIM, HP3);
  }
  // 3. zero padded out_points buffer (cols 144..159 must stay 0 for outproj GEMM)
  ipa_zero_f32<<<nblk((long)NTOK * HP3P, 256), 256, 0, stream>>>(OP, NTOK * HP3P);
  // 4. rigid transforms + fused-QK operand build
  ipa_points<<<nblk((long)NTOK * HNUM, 256), 256, 0, stream>>>(
      Q, K, V, QPr, KPr, VPr, rot, trans, Qc, Kc, Vb, Vpb, r2, c2, scale);
  // 5. pair bias: async-to-LDS double-buffered streaming of 512 MB pair tensor
  {
    long wvs = (long)NTOK * (NTOK / 64);  // 16384 waves, 4 tiles each
    ipa_pairbias<<<nblk(wvs * 32, PB_WAVES * 32), PB_WAVES * 32, 0, stream>>>(
        pair, Wpb, bpb, att);
  }
  // 6. fused scalar+point logits, biases, mask (in place over att)
  {
    long wvs = (long)HNUM * (NTOK / 16) * 4;  // 3072 waves
    ipa_logits<<<nblk(wvs * 32, 256), 256, 0, stream>>>(Qc, Kc, r2, c2, mask, att);
  }
  // 7. softmax rows (48 MB tensor lives in L2)
  ipa_softmax<<<HNUM * NTOK, 256, 0, stream>>>(att);
  // 8. attn @ [V | Vpts]
  {
    long wvs = (long)HNUM * (NTOK / 16);  // 768 waves
    ipa_av<<<nblk(wvs * 32, 256), 256, 0, stream>>>(att, Vb, Vpb, OS, OP);
  }
  // 9. output projection + residual
  {
    long wvs = (long)(NTOK / 16) * (CDIM / 16);
    ipa_outproj<<<nblk(wvs * 32, 256), 256, 0, stream>>>(OS, Wo, bo, OP, Wpo, bpo,
                                                         single, xb);
  }
  // 10. LayerNorm -> d_out
  ipa_layernorm<<<NTOK, 128, 0, stream>>>(xb, gamma, beta, out);
}
